// Network_33260226740732
// MI455X (gfx1250) — matrix operand order repaired and checked by reading, not compiled, not run on hardware
//
#include <hip/hip_runtime.h>
#include <hip/hip_bf16.h>

typedef __attribute__((ext_vector_type(16))) _Float16 v16h;
typedef __attribute__((ext_vector_type(8)))  float    v8f;

#define T_STEPS 15
#define C1 32
#define C2 150
#define HP 131           // padded pooled spike map 131x131
#define HO 130           // conv2 output 130x130
#define NE2 (C2*HO*HO)   // per-timestep conv2 elements = 2,535,000
#define NOUT ((long long)T_STEPS*NE2) // 38,025,000

// ---------------------------------------------------------------- init
__global__ void k_init(unsigned* vmax) {
    if (threadIdx.x == 0 && blockIdx.x == 0) *vmax = 0u;
}

// w2 [150,32,2,2] f32 -> Bt [160][128] f16 (n-major, k contiguous; zero pad n>=150)
__global__ void k_convert_w2(const float* __restrict__ w2, _Float16* __restrict__ Bt) {
    int i = blockIdx.x * 256 + threadIdx.x;
    if (i >= 160 * 128) return;
    int n = i / 128, k = i % 128;
    float v = (n < C2) ? w2[n * 128 + k] : 0.0f;   // k = c*4 + dy*2 + dx matches OIHW flatten
    Bt[i] = (_Float16)v;
}

// ------------------------------------------- conv1 + fire + maxpool + pad
// out: spkA [15][32][131][131] f16 (pooled spikes, zero border = pad1)
__global__ void k_conv1_fire_pool(const float* __restrict__ inp,
                                  const float* __restrict__ w1,
                                  _Float16* __restrict__ spkA) {
    __shared__ float sw[C1 * 2 * 25];
    for (int i = threadIdx.x; i < C1 * 2 * 25; i += blockDim.x) sw[i] = w1[i];
    __syncthreads();

    int idx = blockIdx.x * 256 + threadIdx.x;
    const int tot = T_STEPS * C1 * HP * HP;
    if (idx >= tot) return;
    int px = idx % HP;
    int py = (idx / HP) % HP;
    int c  = (idx / (HP * HP)) % C1;
    int t  =  idx / (HP * HP * C1);

    float out = 0.0f;
    if (px >= 1 && px <= 129 && py >= 1 && py <= 129) {
        int pr = py - 1, pc = px - 1;                 // pooled coords 0..128
        const float* wgt  = &sw[c * 50];
        const float* in_t = inp + (long long)t * 2 * 256 * 256;
        for (int dr = 0; dr < 2; ++dr) {
            int r = 2 * pr - 1 + dr;                  // conv-output row
            if (r < 0 || r > 255) continue;
            for (int dc = 0; dc < 2; ++dc) {
                int cc = 2 * pc - 1 + dc;
                if (cc < 0 || cc > 255) continue;
                float pot = 0.0f;
                for (int ic = 0; ic < 2; ++ic) {
                    const float* ip = in_t + ic * 256 * 256;
                    const float* wp = wgt + ic * 25;
                    #pragma unroll
                    for (int i5 = 0; i5 < 5; ++i5) {
                        int rr = r - 2 + i5;
                        if (rr < 0 || rr > 255) continue;
                        #pragma unroll
                        for (int j5 = 0; j5 < 5; ++j5) {
                            int cj = cc - 2 + j5;
                            if (cj < 0 || cj > 255) continue;
                            pot = fmaf(ip[rr * 256 + cj], wp[i5 * 5 + j5], pot);
                        }
                    }
                }
                if (pot > 10.0f) out = 1.0f;          // fire + pool(max) fused
            }
        }
    }
    spkA[idx] = (_Float16)out;
}

// ------------------------------------------- pointwise inhibition #1
// wave32 per pixel, lane == channel (C1 == 32). In-place on spkA.
__global__ void k_inhibit1(_Float16* __restrict__ spkA) {
    int lane = threadIdx.x & 31;
    int wave = threadIdx.x >> 5;
    int pix  = blockIdx.x * 8 + wave;
    if (pix >= HP * HP) return;
    int h = pix / HP, w = pix % HP;

    float vals[T_STEPS];
    int   mi[T_STEPS];
    int nsp = 0, lastAny = 0;
    #pragma unroll
    for (int t = 0; t < T_STEPS; ++t) {
        float v = (float)spkA[(((t * C1 + lane) * HP) + h) * HP + w];
        vals[t] = v;
        float bv = v; int bi = lane;
        #pragma unroll
        for (int off = 16; off; off >>= 1) {
            float ov = __shfl_xor(bv, off);
            int   oi = __shfl_xor(bi, off);
            if (ov > bv || (ov == bv && oi < bi)) { bv = ov; bi = oi; }
        }
        mi[t] = bi;
        if (bv > 0.0f) nsp++;
        if (t == T_STEPS - 1) lastAny = (bv > 0.0f);
    }
    int e = T_STEPS - nsp; e = e < 0 ? 0 : (e > T_STEPS - 1 ? T_STEPS - 1 : e);
    int win = 0;
    #pragma unroll
    for (int t = 0; t < T_STEPS; ++t) win = (t == e) ? mi[t] : win;
    bool keep = (lane == win) && lastAny;
    #pragma unroll
    for (int t = 0; t < T_STEPS; ++t)
        spkA[(((t * C1 + lane) * HP) + h) * HP + w] =
            keep ? (_Float16)vals[t] : (_Float16)0.0f;
}

// ------------------------------------------- conv2 as WMMA GEMM + fire
// GEMM per (t, y): M = 144 px (9 tiles of 16, 130 valid), N = 160 ch (150 valid),
// K = 128 = 32ch*2*2.  A[m,k] = spkA[t, c, y+dy, x+dx], B[k,n] = w2[n,c,dy,dx].
//
// Block = 320 threads = 10 waves; wave w owns n-tile w, B fragments live in
// registers for the whole block.  A is pre-packed into LDS in fragment-ready
// layout so each lane grabs a full 16xf16 fragment with two ds_load_b128:
//   aF dword index = mt*1024 + s*256 + half*128 + m*8 + j
// holding the f16 pair (dx=0, dx=1) for k = 32*s + kk(j, half).
// A fragments are double-buffered at tile granularity (aT/aN register sets):
// the 8 ds_load_b128 for tile mt+1 issue before the 4 WMMAs of tile mt, so the
// WMMA chain waits only on loads issued a full tile earlier.
__global__ void k_conv2_wmma(const _Float16* __restrict__ spkA,
                             const _Float16* __restrict__ Bt,
                             float* __restrict__ pot2) {
    __shared__ unsigned aF[9 * 1024];                  // 36 KB
    int y = blockIdx.x, t = blockIdx.y;

    int lane = threadIdx.x & 31;
    int wave = threadIdx.x >> 5;                       // 0..9 == n-tile
    int half = lane >> 4, l15 = lane & 15;
    int n = wave * 16 + l15;

    // B fragments: loop-invariant, loaded once (8x global_load_b128 per wave)
    v16h bfr[4];
    #pragma unroll
    for (int s = 0; s < 4; ++s)
        bfr[s] = *(const v16h*)(Bt + n * 128 + 32 * s + 16 * half);

    // prefetch next row of the spike map while we pack this one
    if (threadIdx.x < 64 && y + 2 < HP)
        __builtin_prefetch(spkA + (((t * C1 + (threadIdx.x & 31)) * HP) + (y + 2)) * HP, 0, 1);

    // pack A fragments into LDS (fragment-ready layout)
    const unsigned short* sp = (const unsigned short*)spkA;
    for (int i = threadIdx.x; i < 9 * 1024; i += 320) {
        int j  = i & 7;
        int m  = (i >> 3) & 15;
        int hf = (i >> 7) & 1;
        int s  = (i >> 8) & 3;
        int mt = i >> 10;
        int kk = 16 * hf + 2 * j;
        int k  = 32 * s + kk;                          // even -> dx = 0
        int c  = k >> 2;
        int dy = (k >> 1) & 1;
        int x  = mt * 16 + m;                          // pixel col (0..143)
        const unsigned short* row = sp + (((t * C1 + c) * HP) + (y + dy)) * HP;
        unsigned lo = (x     < HP) ? (unsigned)row[x]     : 0u;  // dx = 0
        unsigned hi = (x + 1 < HP) ? (unsigned)row[x + 1] : 0u;  // dx = 1
        aF[i] = lo | (hi << 16);
    }
    __syncthreads();

    long long rowbase = (((long long)t * C2 + n) * HO + y) * HO;
    int fbase = half * 128 + l15 * 8;                  // per-lane fragment base

    v16h aT[4], aN[4];
    #pragma unroll
    for (int s = 0; s < 4; ++s)                        // tile 0 fragments
        aT[s] = *(const v16h*)(aF + (s * 256 + fbase));

    #pragma unroll
    for (int mt = 0; mt < 9; ++mt) {
        if (mt + 1 < 9) {                              // prefetch whole next tile
            #pragma unroll
            for (int s = 0; s < 4; ++s)
                aN[s] = *(const v16h*)(aF + ((mt + 1) * 1024 + s * 256 + fbase));
        }
        v8f acc = {0.f, 0.f, 0.f, 0.f, 0.f, 0.f, 0.f, 0.f};
        #pragma unroll
        for (int s = 0; s < 4; ++s)
            acc = __builtin_amdgcn_wmma_f32_16x16x32_f16(
                false, aT[s], false, bfr[s], (short)0, acc, false, false);

        // fire(pot2, 1.0) fused into the store; per-lane acc[0..7] are 8
        // consecutive x positions (C/D layout: VGPR r -> M, lane -> N)
        if (n < C2) {
            int xb = mt * 16 + 8 * half;
            if (mt < 8) {                              // statically in-bounds
                #pragma unroll
                for (int r = 0; r < 8; ++r) {
                    float v = acc[r];
                    pot2[rowbase + xb + r] = (v > 1.0f) ? v : 0.0f;
                }
            } else {                                   // tail tile only
                #pragma unroll
                for (int r = 0; r < 8; ++r) {
                    int x = xb + r;
                    if (x < HO) {
                        float v = acc[r];
                        pot2[rowbase + x] = (v > 1.0f) ? v : 0.0f;
                    }
                }
            }
        }
        #pragma unroll
        for (int s = 0; s < 4; ++s) aT[s] = aN[s];     // renamed away by unroll
    }
}

// ------------------------------------------- pointwise inhibition #2 + spk2
__global__ void k_inhibit2(float* __restrict__ pot2, float* __restrict__ spk2) {
    int lane = threadIdx.x & 31;
    int wave = threadIdx.x >> 5;
    int pix  = blockIdx.x * 8 + wave;
    if (pix >= HO * HO) return;
    int h = pix / HO, w = pix % HO;

    int mi[T_STEPS];
    int nsp = 0, lastAny = 0;
    #pragma unroll
    for (int t = 0; t < T_STEPS; ++t) {
        float bv = -1.0f; int bi = 0x7fffffff;
        #pragma unroll
        for (int g = 0; g < 5; ++g) {
            int c = lane + 32 * g;
            if (c < C2) {
                float v = pot2[((((long long)t * C2 + c) * HO) + h) * HO + w];
                if (v > bv || (v == bv && c < bi)) { bv = v; bi = c; }
            }
        }
        #pragma unroll
        for (int off = 16; off; off >>= 1) {
            float ov = __shfl_xor(bv, off);
            int   oi = __shfl_xor(bi, off);
            if (ov > bv || (ov == bv && oi < bi)) { bv = ov; bi = oi; }
        }
        mi[t] = bi;
        if (bv > 0.0f) nsp++;
        if (t == T_STEPS - 1) lastAny = (bv > 0.0f);
    }
    int e = T_STEPS - nsp; e = e < 0 ? 0 : (e > T_STEPS - 1 ? T_STEPS - 1 : e);
    int win = 0;
    #pragma unroll
    for (int t = 0; t < T_STEPS; ++t) win = (t == e) ? mi[t] : win;

    for (int t = 0; t < T_STEPS; ++t)
        #pragma unroll
        for (int g = 0; g < 5; ++g) {
            int c = lane + 32 * g;
            if (c >= C2) continue;
            long long off = ((((long long)t * C2 + c) * HO) + h) * HO + w;
            float v = pot2[off];
            float kv = (c == win && lastAny) ? v : 0.0f;
            pot2[off] = kv;
            spk2[off] = (kv > 0.0f) ? 1.0f : 0.0f;
        }
}

// ------------------------------------------- k-winners: prepare / total / select
__global__ void k_kw_prepare(const float* __restrict__ pot2,
                             const float* __restrict__ spk2,
                             float* __restrict__ wsVal,
                             float* __restrict__ wsNsp,
                             unsigned* __restrict__ vmax) {
    long long i = (long long)blockIdx.x * 256 + threadIdx.x;
    if (i >= NE2) return;
    int nsp = 0;
    #pragma unroll
    for (int t = 0; t < T_STEPS; ++t)
        nsp += (spk2[(long long)t * NE2 + i] > 0.0f) ? 1 : 0;
    int e = T_STEPS - nsp; e = e < 0 ? 0 : (e > T_STEPS - 1 ? T_STEPS - 1 : e);
    float val = pot2[(long long)e * NE2 + i];
    wsVal[i] = val;
    wsNsp[i] = (float)nsp;
    float trunc = (nsp > 0) ? val : 0.0f;              // max_t spk*values == values if spiked
    atomicMax(vmax, __float_as_uint(trunc));           // valid for non-negative floats
}

__global__ void k_kw_total(const float* __restrict__ wsVal,
                           float* __restrict__ wsNsp,   // becomes total in-place
                           const unsigned* __restrict__ vmax) {
    long long i = (long long)blockIdx.x * 256 + threadIdx.x;
    if (i >= NE2) return;
    float v = __uint_as_float(*vmax) * (float)T_STEPS;
    wsNsp[i] = wsNsp[i] * (wsVal[i] + v);
}

__global__ void k_kw_select(float* __restrict__ total, float* __restrict__ winners) {
    __shared__ float smax[1024];
    __shared__ int   sidx[1024];
    for (int k = 0; k < 8; ++k) {
        float bv = -1.0f; int bi = 0x7fffffff;
        for (int i = threadIdx.x; i < NE2; i += 1024) {
            float v = total[i];
            if (v > bv || (v == bv && i < bi)) { bv = v; bi = i; }
        }
        smax[threadIdx.x] = bv; sidx[threadIdx.x] = bi;
        __syncthreads();
        for (int s = 512; s; s >>= 1) {
            if ((int)threadIdx.x < s) {
                float ov = smax[threadIdx.x + s]; int oi = sidx[threadIdx.x + s];
                if (ov > smax[threadIdx.x] ||
                    (ov == smax[threadIdx.x] && oi < sidx[threadIdx.x])) {
                    smax[threadIdx.x] = ov; sidx[threadIdx.x] = oi;
                }
            }
            __syncthreads();
        }
        int   idx  = sidx[0];
        float mval = smax[0];
        int c = idx / (HO * HO);
        int r = (idx % (HO * HO)) / HO;
        int q = idx % HO;
        bool valid = (mval != 0.0f);
        if (threadIdx.x == 0) {
            winners[3 * k + 0] = valid ? (float)c : -1.0f;
            winners[3 * k + 1] = valid ? (float)r : -1.0f;
            winners[3 * k + 2] = valid ? (float)q : -1.0f;
        }
        __syncthreads();
        if (valid) {
            for (int i = threadIdx.x; i < HO * HO; i += 1024)
                total[c * (HO * HO) + i] = 0.0f;                // feature-plane mask
            for (int i = threadIdx.x; i < C2 * 9; i += 1024) {  // 3x3 spatial mask, all ch
                int cc = i / 9, d = i % 9;
                int rr = r + d / 3 - 1, qq = q + d % 3 - 1;
                if (rr >= 0 && rr < HO && qq >= 0 && qq < HO)
                    total[cc * (HO * HO) + rr * HO + qq] = 0.0f;
            }
        }
        __syncthreads();
    }
}

// ---------------------------------------------------------------- launch
extern "C" void kernel_launch(void* const* d_in, const int* in_sizes, int n_in,
                              void* d_out, int out_size, void* d_ws, size_t ws_size,
                              hipStream_t stream) {
    const float* inp = (const float*)d_in[0];   // [15,2,256,256]
    const float* w1  = (const float*)d_in[1];   // [32,2,5,5]
    const float* w2  = (const float*)d_in[2];   // [150,32,2,2]
    (void)d_in[3]; (void)in_sizes; (void)n_in; (void)out_size; (void)ws_size;

    float* pot2 = (float*)d_out;                        // [15,150,130,130]
    float* spk2 = (float*)d_out + NOUT;                 // [15,150,130,130]
    float* wout = (float*)d_out + 2 * NOUT;             // [8,3] winners

    // workspace layout (256B aligned slices)
    char* ws = (char*)d_ws;
    size_t off = 0;
    auto take = [&](size_t bytes) { char* p = ws + off; off += (bytes + 255) & ~(size_t)255; return p; };
    _Float16* spkA  = (_Float16*)take((size_t)T_STEPS * C1 * HP * HP * 2); // 16.5 MB
    _Float16* Bt    = (_Float16*)take((size_t)160 * 128 * 2);              // 40 KB
    float*    wsVal = (float*)take((size_t)NE2 * 4);                       // 10 MB
    float*    wsNsp = (float*)take((size_t)NE2 * 4);                       // 10 MB
    unsigned* vmax  = (unsigned*)take(256);

    k_init<<<1, 64, 0, stream>>>(vmax);
    k_convert_w2<<<(160 * 128 + 255) / 256, 256, 0, stream>>>(w2, Bt);

    int tot1 = T_STEPS * C1 * HP * HP;
    k_conv1_fire_pool<<<(tot1 + 255) / 256, 256, 0, stream>>>(inp, w1, spkA);

    k_inhibit1<<<(HP * HP + 7) / 8, 256, 0, stream>>>(spkA);

    dim3 g2(HO, T_STEPS);                               // one output row per block
    k_conv2_wmma<<<g2, 320, 0, stream>>>(spkA, Bt, pot2);

    k_inhibit2<<<(HO * HO + 7) / 8, 256, 0, stream>>>(pot2, spk2);

    k_kw_prepare<<<(NE2 + 255) / 256, 256, 0, stream>>>(pot2, spk2, wsVal, wsNsp, vmax);
    k_kw_total<<<(NE2 + 255) / 256, 256, 0, stream>>>(wsVal, wsNsp, vmax);
    k_kw_select<<<1, 1024, 0, stream>>>(wsNsp, wout);
}
